// TernaryLinear_87247965651268
// MI455X (gfx1250) — matrix operand order repaired and checked by reading, not compiled, not run on hardware
//
#include <hip/hip_runtime.h>

typedef __attribute__((ext_vector_type(16))) __bf16 v16bf;
typedef __attribute__((ext_vector_type(8)))  __bf16 v8bf;
typedef __attribute__((ext_vector_type(8)))  float  v8f;

#define M_DIM 8192
#define N_DIM 11008
#define K_DIM 4096
#define BM 128
#define BN 128
#define BK 32
#define LDST 40                 // padded LDS row stride (bf16 elems): 80B rows, 16B-aligned chunks
#define KSTEPS (K_DIM / BK)     // 128

union FragU { v16bf v; v8bf h[2]; };

// A fragment (16x32 bf16): half 0 holds K 0..7 & 16..23, half 1 holds K 8..15 & 24..31.
__device__ __forceinline__ v16bf load_frag_a(const __bf16* base, int row, int half) {
    const __bf16* p = base + row * LDST + half * 8;
    FragU u;
    u.h[0] = *reinterpret_cast<const v8bf*>(p);
    u.h[1] = *reinterpret_cast<const v8bf*>(p + 16);
    return u.v;
}

// B fragment (32x16 bf16): half 0 holds K 0..7 & 16..23, half 1 holds K 8..15 & 24..31.
__device__ __forceinline__ v16bf load_frag_b(const __bf16* base, int row, int half) {
    const __bf16* p = base + row * LDST + half * 8;
    FragU u;
    u.h[0] = *reinterpret_cast<const v8bf*>(p);
    u.h[1] = *reinterpret_cast<const v8bf*>(p + 16);
    return u.v;
}

__global__ __launch_bounds__(256)
void ternary_linear_wmma(const float* __restrict__ x,
                         const float* __restrict__ w,
                         const float* __restrict__ bias,
                         float* __restrict__ out) {
    // Double-buffered stages: 2 * (128*40 + 128*40) * 2B = 40 KB (of 320 KB/WGP).
    __shared__ __align__(16) __bf16 ldsA[2][BM * LDST];
    __shared__ __align__(16) __bf16 ldsB[2][BN * LDST];

    const int tid = threadIdx.x;
    const int m0  = blockIdx.y * BM;    // token-block base
    const int n0  = blockIdx.x * BN;    // out-feature-block base

    // Global->register staging: 2 threads per 32-float row, 16 floats each.
    const int grow  = tid >> 1;         // 0..127
    const int gpart = tid & 1;

    // 8 waves -> 4 (M) x 2 (N); each wave owns a 32x64 tile = 2x4 WMMA accumulators.
    const int lane = tid & 31;
    const int wave = tid >> 5;
    const int wm   = wave >> 1;         // 0..3
    const int wn   = wave & 1;          // 0..1
    const int r    = lane & 15;
    const int half = lane >> 4;

    v8f acc[2][4];
    const v8f vzero = {0.f, 0.f, 0.f, 0.f, 0.f, 0.f, 0.f, 0.f};
#pragma unroll
    for (int mt = 0; mt < 2; ++mt)
#pragma unroll
        for (int nt = 0; nt < 4; ++nt)
            acc[mt][nt] = vzero;

    const float* gA = x + (size_t)(m0 + grow) * K_DIM + gpart * 16;
    const float* gB = w + (size_t)(n0 + grow) * K_DIM + gpart * 16;
    const int soff = grow * LDST + gpart * 16;

    float aR[16], bR[16];

    // ---- stage 0: global -> regs -> convert/ternarize -> LDS buf 0 ----
#pragma unroll
    for (int c = 0; c < 4; ++c) {
        float4 va = *reinterpret_cast<const float4*>(gA + c * 4);
        float4 vb = *reinterpret_cast<const float4*>(gB + c * 4);
        aR[c*4+0] = va.x; aR[c*4+1] = va.y; aR[c*4+2] = va.z; aR[c*4+3] = va.w;
        bR[c*4+0] = vb.x; bR[c*4+1] = vb.y; bR[c*4+2] = vb.z; bR[c*4+3] = vb.w;
    }
    {
        __bf16 ta[16], tb[16];
#pragma unroll
        for (int i = 0; i < 16; ++i) {
            ta[i] = (__bf16)aR[i];
            float wv = bR[i];
            float s  = (wv > 0.f) ? 1.f : ((wv < 0.f) ? -1.f : 0.f);
            tb[i] = (__bf16)s;
        }
        *reinterpret_cast<v8bf*>(&ldsA[0][soff])     = *reinterpret_cast<const v8bf*>(&ta[0]);
        *reinterpret_cast<v8bf*>(&ldsA[0][soff + 8]) = *reinterpret_cast<const v8bf*>(&ta[8]);
        *reinterpret_cast<v8bf*>(&ldsB[0][soff])     = *reinterpret_cast<const v8bf*>(&tb[0]);
        *reinterpret_cast<v8bf*>(&ldsB[0][soff + 8]) = *reinterpret_cast<const v8bf*>(&tb[8]);
    }
    __syncthreads();

    for (int kt = 0; kt < KSTEPS; ++kt) {
        const int cur = kt & 1;
        const int nxt = cur ^ 1;

        // ---- issue next stage's global loads (latency hidden by the WMMA burst) ----
        if (kt + 1 < KSTEPS) {
            const float* pA = gA + (size_t)(kt + 1) * BK;
            const float* pB = gB + (size_t)(kt + 1) * BK;
#pragma unroll
            for (int c = 0; c < 4; ++c) {
                float4 va = *reinterpret_cast<const float4*>(pA + c * 4);
                float4 vb = *reinterpret_cast<const float4*>(pB + c * 4);
                aR[c*4+0] = va.x; aR[c*4+1] = va.y; aR[c*4+2] = va.z; aR[c*4+3] = va.w;
                bR[c*4+0] = vb.x; bR[c*4+1] = vb.y; bR[c*4+2] = vb.z; bR[c*4+3] = vb.w;
            }
        }

        // ---- compute from current stage: 2 A frags, 4 B frags, 8 WMMAs ----
        v16bf af[2], bfr[4];
#pragma unroll
        for (int mt = 0; mt < 2; ++mt)
            af[mt] = load_frag_a(ldsA[cur], wm * 32 + mt * 16 + r, half);
#pragma unroll
        for (int nt = 0; nt < 4; ++nt)
            bfr[nt] = load_frag_b(ldsB[cur], wn * 64 + nt * 16 + r, half);

#pragma unroll
        for (int mt = 0; mt < 2; ++mt)
#pragma unroll
            for (int nt = 0; nt < 4; ++nt)
                acc[mt][nt] = __builtin_amdgcn_wmma_f32_16x16x32_bf16(
                    false, af[mt], false, bfr[nt],
                    (short)0, acc[mt][nt], false, false);

        // ---- convert/ternarize + store into the other stage (no conflict with reads) ----
        if (kt + 1 < KSTEPS) {
            __bf16 ta[16], tb[16];
#pragma unroll
            for (int i = 0; i < 16; ++i) {
                ta[i] = (__bf16)aR[i];
                float wv = bR[i];
                float s  = (wv > 0.f) ? 1.f : ((wv < 0.f) ? -1.f : 0.f);
                tb[i] = (__bf16)s;
            }
            *reinterpret_cast<v8bf*>(&ldsA[nxt][soff])     = *reinterpret_cast<const v8bf*>(&ta[0]);
            *reinterpret_cast<v8bf*>(&ldsA[nxt][soff + 8]) = *reinterpret_cast<const v8bf*>(&ta[8]);
            *reinterpret_cast<v8bf*>(&ldsB[nxt][soff])     = *reinterpret_cast<const v8bf*>(&tb[0]);
            *reinterpret_cast<v8bf*>(&ldsB[nxt][soff + 8]) = *reinterpret_cast<const v8bf*>(&tb[8]);
        }

        // Single barrier: orders this stage's reads before next iter's overwrite,
        // and next stage's stores before next iter's reads.
        __syncthreads();
    }

    // ---- epilogue: bias add + non-temporal stores (output is streamed once;
    //      keep the reused x/W lines resident in the 192 MB L2) ----
    float bv[4];
#pragma unroll
    for (int nt = 0; nt < 4; ++nt)
        bv[nt] = bias[n0 + wn * 64 + nt * 16 + r];

#pragma unroll
    for (int mt = 0; mt < 2; ++mt) {
        const int rowb = m0 + wm * 32 + mt * 16 + half * 8;
#pragma unroll
        for (int nt = 0; nt < 4; ++nt) {
            const int col = n0 + wn * 64 + nt * 16 + r;
#pragma unroll
            for (int i = 0; i < 8; ++i)
                __builtin_nontemporal_store(acc[mt][nt][i] + bv[nt],
                                            &out[(size_t)(rowb + i) * N_DIM + col]);
        }
    }
}

extern "C" void kernel_launch(void* const* d_in, const int* in_sizes, int n_in,
                              void* d_out, int out_size, void* d_ws, size_t ws_size,
                              hipStream_t stream) {
    const float* x    = (const float*)d_in[0];
    const float* w    = (const float*)d_in[1];
    const float* bias = (const float*)d_in[2];
    float* out = (float*)d_out;

    dim3 grid(N_DIM / BN, M_DIM / BM);   // (86, 64)
    ternary_linear_wmma<<<grid, 256, 0, stream>>>(x, w, bias, out);
}
